// MambaBlock_58909771432039
// MI455X (gfx1250) — matrix operand order repaired and checked by reading, not compiled, not run on hardware
//
#include <hip/hip_runtime.h>
#include <math.h>

// ---------------------------------------------------------------------------
// Mamba block, MI455X (gfx1250, wave32).
//  * Dominant GEMMs (x*W_in^T, y*W_out^T): split-precision bf16x3 on
//    V_WMMA_F32_16X16X32_BF16, 2x2 register-blocked (32x32 per wave) so each
//    K=32 step does 12 WMMAs on 8 fragment loads (2x fewer loads per WMMA).
//  * Small GEMMs (W_x proj N=96, dt proj K=64): exact fp32 WMMA 16x16x4.
//  * Selective scan fused and sequential in L (clip => non-associative).
// ---------------------------------------------------------------------------

typedef float  v2f   __attribute__((ext_vector_type(2)));
typedef float  v8f   __attribute__((ext_vector_type(8)));
typedef __bf16 v8bf  __attribute__((ext_vector_type(8)));
typedef __bf16 v16bf __attribute__((ext_vector_type(16)));

#define D_MODEL 1024
#define D_STATE 16
#define D_INNER 2048
#define DT_RANK 64
#define NTOK    2048   // B * L
#define LSEQ    1024
#define XDBL_LD 96     // DT_RANK + 2*D_STATE

// ---------------------------------------------------------------------------
// fp32 NT GEMM: C[M,N] = A[M,K] * W[N,K]^T, one wave per 16x16 tile.
//   mode 0: plain store
//   mode 1: delta epilogue: v = c + bias[n]; v = min(softplus(v)+1e-4, 10)
// ---------------------------------------------------------------------------
__global__ void wmma_gemm_nt_f32(const float* __restrict__ A, int lda,
                                 const float* __restrict__ W, int ldb,
                                 float* __restrict__ C, int ldc,
                                 int tilesN, int K, int mode,
                                 const float* __restrict__ bias)
{
    const int wave = threadIdx.x >> 5;
    const int lane = threadIdx.x & 31;
    const int tile = blockIdx.x * (blockDim.x >> 5) + wave;
    const int tm = tile / tilesN;
    const int tn = tile % tilesN;

    const int half = lane >> 4;
    const int r    = lane & 15;

    const float* ap = A + (size_t)(tm * 16 + r) * lda + 2 * half;
    const float* bp = W + (size_t)(tn * 16 + r) * ldb + 2 * half;

    v8f c = {};
#pragma unroll 8
    for (int k = 0; k < K; k += 4) {
        v2f a = *(const v2f*)(ap + k);
        v2f b = *(const v2f*)(bp + k);
        c = __builtin_amdgcn_wmma_f32_16x16x4_f32(false, a, false, b,
                                                  (short)0, c, false, false);
    }

    float* cp = C + (size_t)(tm * 16 + 8 * half) * ldc + tn * 16 + r;
    if (mode == 0) {
#pragma unroll
        for (int rr = 0; rr < 8; ++rr)
            cp[(size_t)rr * ldc] = c[rr];
    } else {
        const float bn = bias[tn * 16 + r];
#pragma unroll
        for (int rr = 0; rr < 8; ++rr) {
            float v = c[rr] + bn;
            float sp = (v > 20.0f) ? v : log1pf(expf(v));
            cp[(size_t)rr * ldc] = fminf(sp + 1e-4f, 10.0f);
        }
    }
}

// ---------------------------------------------------------------------------
// Split fp32 -> (bf16 hi, bf16 lo) with lo = bf16(x - float(hi)).
// ---------------------------------------------------------------------------
__global__ void split_bf16_kernel(const float* __restrict__ src,
                                  __bf16* __restrict__ hi,
                                  __bf16* __restrict__ lo)
{
    const int i = blockIdx.x * blockDim.x + threadIdx.x;
    const float v = src[i];
    const __bf16 h = (__bf16)v;
    hi[i] = h;
    lo[i] = (__bf16)(v - (float)h);
}

// ---------------------------------------------------------------------------
// bf16 fragment loads (ISA 7.12.2, 16-bit A 16x32 / B 32x16):
//   A lane(half,r): VGPR0-3 = K 8h..8h+7 ; VGPR4-7 = K 16+8h..+7
//   B lane(half,n): VGPR0-3 = K 8h..8h+7 ; VGPR4-7 = K 16+8h..+7 (same as A)
// Both contiguous in K for row-major K-innermost storage -> b128 loads.
// ---------------------------------------------------------------------------
__device__ __forceinline__ v16bf load_a_frag(const __bf16* row, int k0, int half)
{
    v8bf a0 = *(const v8bf*)(row + k0 + 8 * half);
    v8bf a1 = *(const v8bf*)(row + k0 + 16 + 8 * half);
    return __builtin_shufflevector(a0, a1, 0, 1, 2, 3, 4, 5, 6, 7,
                                   8, 9, 10, 11, 12, 13, 14, 15);
}

__device__ __forceinline__ v16bf load_b_frag(const __bf16* row, int k0, int half)
{
    v8bf b0 = *(const v8bf*)(row + k0 + 8 * half);
    v8bf b1 = *(const v8bf*)(row + k0 + 16 + 8 * half);
    return __builtin_shufflevector(b0, b1, 0, 1, 2, 3, 4, 5, 6, 7,
                                   8, 9, 10, 11, 12, 13, 14, 15);
}

__device__ __forceinline__ v8f wmma_bf16(v16bf a, v16bf b, v8f c)
{
    return __builtin_amdgcn_wmma_f32_16x16x32_bf16(false, a, false, b,
                                                   (short)0, c, false, false);
}

// ---------------------------------------------------------------------------
// bf16x3 NT GEMM, 2x2 register-blocked: each wave computes a 32x32 C block.
// C = Ahi*Bhi^T + Ahi*Blo^T + Alo*Bhi^T, fp32 accumulate.
// Per K=32 step: 8 fragments (16 b128 loads) -> 12 WMMAs, 4 indep acc chains.
// ---------------------------------------------------------------------------
__global__ void wmma_gemm_nt_bf16x3_2x2(const __bf16* __restrict__ Ahi,
                                        const __bf16* __restrict__ Alo, int lda,
                                        const __bf16* __restrict__ Bhi,
                                        const __bf16* __restrict__ Blo, int ldb,
                                        float* __restrict__ C, int ldc,
                                        int tilesN2, int K)
{
    const int wave = threadIdx.x >> 5;
    const int lane = threadIdx.x & 31;
    const int tile = blockIdx.x * (blockDim.x >> 5) + wave;
    const int tm = tile / tilesN2;          // 32-row macro tile
    const int tn = tile % tilesN2;          // 32-col macro tile

    const int half = lane >> 4;
    const int r    = lane & 15;
    const int m0 = tm * 32;
    const int n0 = tn * 32;

    const __bf16* ah0 = Ahi + (size_t)(m0 + r) * lda;
    const __bf16* al0 = Alo + (size_t)(m0 + r) * lda;
    const __bf16* ah1 = ah0 + (size_t)16 * lda;
    const __bf16* al1 = al0 + (size_t)16 * lda;
    const __bf16* bh0 = Bhi + (size_t)(n0 + r) * ldb;
    const __bf16* bl0 = Blo + (size_t)(n0 + r) * ldb;
    const __bf16* bh1 = bh0 + (size_t)16 * ldb;
    const __bf16* bl1 = bl0 + (size_t)16 * ldb;

    v8f c00 = {}, c01 = {}, c10 = {}, c11 = {};
#pragma unroll 2
    for (int k = 0; k < K; k += 32) {
        v16bf Ah0 = load_a_frag(ah0, k, half);
        v16bf Al0 = load_a_frag(al0, k, half);
        v16bf Ah1 = load_a_frag(ah1, k, half);
        v16bf Al1 = load_a_frag(al1, k, half);
        v16bf Bh0 = load_b_frag(bh0, k, half);
        v16bf Bl0 = load_b_frag(bl0, k, half);
        v16bf Bh1 = load_b_frag(bh1, k, half);
        v16bf Bl1 = load_b_frag(bl1, k, half);

        c00 = wmma_bf16(Ah0, Bh0, c00);
        c01 = wmma_bf16(Ah0, Bh1, c01);
        c10 = wmma_bf16(Ah1, Bh0, c10);
        c11 = wmma_bf16(Ah1, Bh1, c11);
        c00 = wmma_bf16(Ah0, Bl0, c00);
        c01 = wmma_bf16(Ah0, Bl1, c01);
        c10 = wmma_bf16(Ah1, Bl0, c10);
        c11 = wmma_bf16(Ah1, Bl1, c11);
        c00 = wmma_bf16(Al0, Bh0, c00);
        c01 = wmma_bf16(Al0, Bh1, c01);
        c10 = wmma_bf16(Al1, Bh0, c10);
        c11 = wmma_bf16(Al1, Bh1, c11);
    }

    float* cp0 = C + (size_t)(m0 + 8 * half) * ldc + n0 + r;        // tiles (0,*)
    float* cp1 = C + (size_t)(m0 + 16 + 8 * half) * ldc + n0 + r;   // tiles (1,*)
#pragma unroll
    for (int rr = 0; rr < 8; ++rr) {
        cp0[(size_t)rr * ldc]      = c00[rr];
        cp0[(size_t)rr * ldc + 16] = c01[rr];
        cp1[(size_t)rr * ldc]      = c10[rr];
        cp1[(size_t)rr * ldc + 16] = c11[rr];
    }
}

// ---------------------------------------------------------------------------
// Causal depthwise conv (width 4) + bias + SiLU over x-path half of xz.
// ---------------------------------------------------------------------------
__global__ void conv_silu_kernel(const float* __restrict__ xz,
                                 const float* __restrict__ conv_w,
                                 const float* __restrict__ conv_b,
                                 float* __restrict__ xc)
{
    const int idx = blockIdx.x * blockDim.x + threadIdx.x;
    const int d = idx & (D_INNER - 1);
    const int t = idx >> 11;
    const int b = t >> 10;
    const int l = t & (LSEQ - 1);

    float acc = conv_b[d];
#pragma unroll
    for (int j = 0; j < 4; ++j) {
        const int ll = l - 3 + j;
        if (ll >= 0)
            acc += conv_w[d * 4 + j] *
                   xz[(size_t)((b << 10) + ll) * (2 * D_INNER) + d];
    }
    const float s = acc / (1.0f + expf(-acc));
    xc[(size_t)t * D_INNER + d] = s;
}

// ---------------------------------------------------------------------------
// Fused selective scan: one thread per (batch, channel); sequential in L.
// ---------------------------------------------------------------------------
__global__ void scan_kernel(const float* __restrict__ delta,
                            const float* __restrict__ xdbl,
                            const float* __restrict__ xc,
                            const float* __restrict__ xz,
                            const float* __restrict__ A_log,
                            const float* __restrict__ D_param,
                            float* __restrict__ y)
{
    const int gid = blockIdx.x * blockDim.x + threadIdx.x;
    const int b = gid >> 11;
    const int d = gid & (D_INNER - 1);

    float A[D_STATE];
#pragma unroll
    for (int n = 0; n < D_STATE; ++n)
        A[n] = -expf(A_log[d * D_STATE + n]);
    const float Dp = D_param[d];

    float h[D_STATE];
#pragma unroll
    for (int n = 0; n < D_STATE; ++n) h[n] = 0.0f;

    for (int l = 0; l < LSEQ; ++l) {
        const int t = (b << 10) + l;
        const float dt  = delta[(size_t)t * D_INNER + d];
        const float xct = xc[(size_t)t * D_INNER + d];
        const float zt  = xz[(size_t)t * (2 * D_INNER) + D_INNER + d];
        const float* bc = xdbl + (size_t)t * XDBL_LD;

        float yacc = 0.0f;
#pragma unroll
        for (int n = 0; n < D_STATE; ++n) {
            const float Bn = bc[DT_RANK + n];
            const float Cn = bc[DT_RANK + D_STATE + n];
            float a = fminf(expf(dt * A[n]), 10.0f);
            float bu = dt * Bn * xct;
            bu = fminf(fmaxf(bu, -10.0f), 10.0f);
            float hn = a * h[n] + bu;
            hn = fminf(fmaxf(hn, -20.0f), 20.0f);
            h[n] = hn;
            yacc += hn * Cn;
        }
        float yv = yacc + xct * Dp;
        yv = fminf(fmaxf(yv, -50.0f), 50.0f);
        const float sz = zt / (1.0f + expf(-zt));
        y[(size_t)t * D_INNER + d] = yv * sz;
    }
}

// ---------------------------------------------------------------------------
// Host launcher. Workspace (float offsets), total 21,168,128 floats (~84.7MB):
//   xz    [0,         8388608)   -- stage-6 bf16 area aliases here (24MB<=32MB,
//                                   xz dead after scan consumes z-path)
//   xc    [8388608,  12582912)   -- stage-1 bf16 area aliases [8388608,14680064)
//   xdbl  [12582912, 12779520)      (written only after stage-1 GEMM finishes)
//   delta [12779520, 16973824)
//   y     [16973824, 21168128)
// ---------------------------------------------------------------------------
extern "C" void kernel_launch(void* const* d_in, const int* in_sizes, int n_in,
                              void* d_out, int out_size, void* d_ws, size_t ws_size,
                              hipStream_t stream)
{
    const float* x       = (const float*)d_in[0];
    const float* W_in    = (const float*)d_in[1];
    const float* conv_w  = (const float*)d_in[2];
    const float* conv_b  = (const float*)d_in[3];
    const float* W_x     = (const float*)d_in[4];
    const float* W_dt    = (const float*)d_in[5];
    const float* b_dt    = (const float*)d_in[6];
    const float* A_log   = (const float*)d_in[7];
    const float* D_param = (const float*)d_in[8];
    const float* W_out   = (const float*)d_in[9];
    float* out = (float*)d_out;

    float* ws    = (float*)d_ws;
    float* xz    = ws;
    float* xc    = ws + (size_t)NTOK * 2 * D_INNER;
    float* xdbl  = xc + (size_t)NTOK * D_INNER;
    float* delta = xdbl + (size_t)NTOK * XDBL_LD;
    float* y     = delta + (size_t)NTOK * D_INNER;

    // Stage-1 bf16 split buffers alias the xc/xdbl/delta region (dead now).
    __bf16* s1    = (__bf16*)xc;
    __bf16* x_hi  = s1;                                     // 2M bf16
    __bf16* x_lo  = x_hi + (size_t)NTOK * D_MODEL;          // 2M
    __bf16* wi_hi = x_lo + (size_t)NTOK * D_MODEL;          // 4M
    __bf16* wi_lo = wi_hi + (size_t)2 * D_INNER * D_MODEL;  // 4M  (24MB total)

    // Stage-6 bf16 split buffers alias the xz region (dead after scan).
    __bf16* s6    = (__bf16*)xz;
    __bf16* y_hi  = s6;                                     // 4M bf16
    __bf16* y_lo  = y_hi + (size_t)NTOK * D_INNER;          // 4M
    __bf16* wo_hi = y_lo + (size_t)NTOK * D_INNER;          // 2M
    __bf16* wo_lo = wo_hi + (size_t)D_MODEL * D_INNER;      // 2M  (24MB total)

    // --- 1) xz = x * W_in^T  (bf16x3, 2x2)   M=2048 N=4096 K=1024
    split_bf16_kernel<<<(NTOK * D_MODEL) / 256, 256, 0, stream>>>(x, x_hi, x_lo);
    split_bf16_kernel<<<(2 * D_INNER * D_MODEL) / 256, 256, 0, stream>>>(W_in, wi_hi, wi_lo);
    wmma_gemm_nt_bf16x3_2x2<<<(64 * 128) / 8, 256, 0, stream>>>(
        x_hi, x_lo, D_MODEL, wi_hi, wi_lo, D_MODEL, xz, 2 * D_INNER,
        /*tilesN2=*/128, /*K=*/D_MODEL);

    // --- 2) depthwise conv + SiLU  (overwrites the stage-1 bf16 area)
    conv_silu_kernel<<<(NTOK * D_INNER) / 256, 256, 0, stream>>>(
        xz, conv_w, conv_b, xc);

    // --- 3) x_dbl = xc * W_x^T  (fp32 WMMA)   M=2048 N=96 K=2048
    wmma_gemm_nt_f32<<<768 / 8, 256, 0, stream>>>(
        xc, D_INNER, W_x, D_INNER, xdbl, XDBL_LD,
        XDBL_LD / 16, D_INNER, /*mode=*/0, nullptr);

    // --- 4) delta = min(softplus(x_dbl[:,:64]*W_dt^T + b_dt)+1e-4, 10)
    //         (fp32 WMMA)                      M=2048 N=2048 K=64
    wmma_gemm_nt_f32<<<16384 / 8, 256, 0, stream>>>(
        xdbl, XDBL_LD, W_dt, DT_RANK, delta, D_INNER,
        D_INNER / 16, DT_RANK, /*mode=*/1, b_dt);

    // --- 5) fused selective scan -> y (gated); last reader of xz (z-path)
    scan_kernel<<<16, 256, 0, stream>>>(delta, xdbl, xc, xz, A_log, D_param, y);

    // --- 6) out = y * W_out^T  (bf16x3, 2x2)  M=2048 N=1024 K=2048
    split_bf16_kernel<<<(NTOK * D_INNER) / 256, 256, 0, stream>>>(y, y_hi, y_lo);
    split_bf16_kernel<<<(D_MODEL * D_INNER) / 256, 256, 0, stream>>>(W_out, wo_hi, wo_lo);
    wmma_gemm_nt_bf16x3_2x2<<<(64 * 32) / 8, 256, 0, stream>>>(
        y_hi, y_lo, D_INNER, wo_hi, wo_lo, D_INNER, out, D_MODEL,
        /*tilesN2=*/32, /*K=*/D_INNER);
}
